// BlockAttnRes_90537910600427
// MI455X (gfx1250) — compile-verified, matrix operand order checked by reading, not run on hardware
//
#include <hip/hip_runtime.h>
#include <math.h>

// =====================================================================
// BlockAttnRes for MI455X (gfx1250, wave32, WMMA bf16 w/ f32 accum)
//
//  - fused_layer_kernel : mix(softmax over n<=5 sources) -> LayerNorm ->
//        GEMM1(128x256) -> GELU -> GEMM2(256x128) -> partial accumulate,
//        128 tokens per workgroup (8 waves, one 16-row M tile per wave),
//        v_wmma_f32_16x16x32_bf16 with f32 accumulation.
//        h never touches HBM (registers -> LDS only).
//  - mix_kernel  : layer 15 only (output = h, its MLP is dead code).
//  - prep_kernel : one-time f32 -> bf16 weight repack into WMMA
//        B-fragment tiling (per fragment: 32 lanes x 16 halves contiguous
//        so each lane's fragment is two coalesced global_load_b128).
//  GELU uses the branchless sigmoid/tanh form (error << bf16 rounding),
//  keeping the GEMM loop bodies small enough for full unroll + load
//  pipelining, and avoiding EXEC-divergent OCML erf.
// =====================================================================

#define LNUM 16
#define DDIM 128
#define TILE_M 128        // tokens per workgroup in fused kernel

typedef __bf16 bf16x16 __attribute__((ext_vector_type(16)));
typedef float  f32x8   __attribute__((ext_vector_type(8)));

struct FragU { uint4 lo, hi; };   // 32 bytes == 16 bf16 halves

static __device__ __forceinline__ bf16x16 as_bf(FragU f) {
    return __builtin_bit_cast(bf16x16, f);
}

// round-to-nearest-even f32 -> bf16 (raw u16)
static __device__ __forceinline__ unsigned short f2bf(float f) {
    unsigned int u = __float_as_uint(f);
    u += 0x7FFFu + ((u >> 16) & 1u);
    return (unsigned short)(u >> 16);
}

// branchless GELU: 0.5x(1+tanh(0.79788456(x+0.044715x^3))) = x*sigmoid(2u)
static __device__ __forceinline__ float fast_gelu(float x) {
    float u  = 1.5957691216f * (x + 0.044715f * x * x * x);   // 2 * sqrt(2/pi) * (...)
    float ex = __expf(-u);                                    // v_exp_f32 path
    return x * __frcp_rn(1.0f + ex);
}

// ---------------------------------------------------------------------
// Weight repack: src [L][K][N] f32 row-major -> dst bf16 fragment-tiled.
// Half h of lane L in fragment (kt,nt):
//   j = h/2, ps = h%2
//   k = kt*32 + (j<4 ? 2j : 16+2(j-4)) + (L>=16 ? 8 : 0) + ps
//   n = nt*16 + (L&15)
// => halves 0..7 hold K[base..base+7], halves 8..15 hold K[base+16..+23],
//    lanes 16..31 get the +8 K offset (ISA 16-bit A/B VGPR packing).
// ---------------------------------------------------------------------
__global__ __launch_bounds__(256)
void prep_kernel(const float* __restrict__ src, unsigned short* __restrict__ dst,
                 int K, int N, int NNT, long total)
{
    long tid = (long)blockIdx.x * blockDim.x + threadIdx.x;
    if (tid >= total) return;
    long per   = (long)K * N;
    int  layer = (int)(tid / per);
    int  rem   = (int)(tid % per);
    int  hh    = rem & 15;
    int  lane  = (rem >> 4) & 31;
    int  nt    = (rem >> 9) % NNT;
    int  kt    = rem / (512 * NNT);
    int  j  = hh >> 1, ps = hh & 1;
    int  k  = kt * 32 + ((j < 4) ? 2 * j : 16 + 2 * (j - 4)) + ((lane >= 16) ? 8 : 0) + ps;
    int  n  = nt * 16 + (lane & 15);
    dst[tid] = f2bf(src[(long)layer * per + (long)k * N + n]);
}

// ---------------------------------------------------------------------
// Standalone mixing (layer 15 only): one wave per token -> d_out.
// ---------------------------------------------------------------------
__global__ __launch_bounds__(256)
void mix_kernel(const float* s0, const float* s1, const float* s2,
                const float* s3, const float* s4,
                int n, const float* __restrict__ wl,
                float* __restrict__ out, long ntok)
{
    const int  lane  = threadIdx.x & 31;
    const long token = (long)blockIdx.x * 8 + (threadIdx.x >> 5);
    if (token >= ntok) return;
    const long  base = token * DDIM + lane * 4;
    const float* srcs[5] = { s0, s1, s2, s3, s4 };

    const float4 w4 = *(const float4*)(wl + lane * 4);
    float4 v[5];
    float  logit[5];
    #pragma unroll
    for (int i = 0; i < 5; ++i) {
        if (i >= n) break;
        float4 x = *(const float4*)(srcs[i] + base);
        v[i] = x;
        float ss = x.x * x.x + x.y * x.y + x.z * x.z + x.w * x.w;
        float dp = w4.x * x.x + w4.y * x.y + w4.z * x.z + w4.w * x.w;
        #pragma unroll
        for (int m = 16; m >= 1; m >>= 1) {
            ss += __shfl_xor(ss, m, 32);
            dp += __shfl_xor(dp, m, 32);
        }
        logit[i] = dp * rsqrtf(ss * (1.0f / 128.0f) + 1e-8f);
    }
    float mx = logit[0];
    for (int i = 1; i < n; ++i) mx = fmaxf(mx, logit[i]);
    float den = 0.f, we[5];
    for (int i = 0; i < n; ++i) { we[i] = __expf(logit[i] - mx); den += we[i]; }
    const float inv = 1.0f / den;
    float4 acc = make_float4(0.f, 0.f, 0.f, 0.f);
    for (int i = 0; i < n; ++i) {
        float c = we[i] * inv;
        acc.x += c * v[i].x; acc.y += c * v[i].y;
        acc.z += c * v[i].z; acc.w += c * v[i].w;
    }
    *(float4*)(out + base) = acc;
}

// ---------------------------------------------------------------------
// Fused layer: mix -> LN -> GEMM1 -> GELU -> GEMM2 -> partial accumulate.
// 128 tokens x D=128 per workgroup; wave w owns M-tile w (16 rows).
// ---------------------------------------------------------------------
__global__ __launch_bounds__(256)
void fused_layer_kernel(const float* s0, const float* s1, const float* s2,
                        const float* s3, const float* s4, int n,
                        const float* __restrict__ wl,
                        const float* __restrict__ gvec, const float* __restrict__ bvec,
                        const unsigned short* __restrict__ w1t, const float* __restrict__ b1,
                        const unsigned short* __restrict__ w2t, const float* __restrict__ b2,
                        float* __restrict__ outp, int accumulate)
{
    __shared__ __align__(16) unsigned short Xa[TILE_M * 128];   // LN(h) bf16
    __shared__ __align__(16) unsigned short Hb[TILE_M * 256];   // GELU(h1) bf16

    const int  tid     = threadIdx.x;
    const int  wv      = tid >> 5;
    const int  lane    = tid & 31;
    const long tokBase = (long)blockIdx.x * TILE_M;
    const float* srcs[5] = { s0, s1, s2, s3, s4 };

    // ---- Phase 1: mix + LayerNorm, h stays on-chip (regs -> LDS) ----
    const float4 w4  = *(const float4*)(wl   + lane * 4);
    const float4 g4  = *(const float4*)(gvec + lane * 4);
    const float4 bl4 = *(const float4*)(bvec + lane * 4);
    for (int r = 0; r < 16; ++r) {
        const int  row  = wv * 16 + r;
        const long base = (tokBase + row) * DDIM + lane * 4;

        float4 v[5];
        float  logit[5];
        #pragma unroll
        for (int i = 0; i < 5; ++i) {
            if (i >= n) break;
            float4 x = *(const float4*)(srcs[i] + base);
            v[i] = x;
            float ss = x.x * x.x + x.y * x.y + x.z * x.z + x.w * x.w;
            float dp = w4.x * x.x + w4.y * x.y + w4.z * x.z + w4.w * x.w;
            #pragma unroll
            for (int m = 16; m >= 1; m >>= 1) {
                ss += __shfl_xor(ss, m, 32);
                dp += __shfl_xor(dp, m, 32);
            }
            logit[i] = dp * rsqrtf(ss * (1.0f / 128.0f) + 1e-8f);
        }
        float mx = logit[0];
        for (int i = 1; i < n; ++i) mx = fmaxf(mx, logit[i]);
        float den = 0.f, we[5];
        for (int i = 0; i < n; ++i) { we[i] = __expf(logit[i] - mx); den += we[i]; }
        const float winv = 1.0f / den;
        float4 h = make_float4(0.f, 0.f, 0.f, 0.f);
        for (int i = 0; i < n; ++i) {
            float c = we[i] * winv;
            h.x += c * v[i].x; h.y += c * v[i].y;
            h.z += c * v[i].z; h.w += c * v[i].w;
        }

        // LayerNorm on h (row stats via wave32 xor reduce)
        float s = h.x + h.y + h.z + h.w;
        float q = h.x * h.x + h.y * h.y + h.z * h.z + h.w * h.w;
        #pragma unroll
        for (int m = 16; m >= 1; m >>= 1) {
            s += __shfl_xor(s, m, 32);
            q += __shfl_xor(q, m, 32);
        }
        const float mu   = s * (1.0f / 128.0f);
        const float var  = q * (1.0f / 128.0f) - mu * mu;
        const float rinv = rsqrtf(var + 1e-5f);
        float4 xn;
        xn.x = (h.x - mu) * rinv * g4.x + bl4.x;
        xn.y = (h.y - mu) * rinv * g4.y + bl4.y;
        xn.z = (h.z - mu) * rinv * g4.z + bl4.z;
        xn.w = (h.w - mu) * rinv * g4.w + bl4.w;
        unsigned int p0 = (unsigned int)f2bf(xn.x) | ((unsigned int)f2bf(xn.y) << 16);
        unsigned int p1 = (unsigned int)f2bf(xn.z) | ((unsigned int)f2bf(xn.w) << 16);
        *(uint2*)&Xa[row * 128 + lane * 4] = make_uint2(p0, p1);
    }
    __syncthreads();

    const int mt     = wv;                      // M tile 0..7, unique per wave
    const int rowoff = (lane >> 4) << 3;        // 0 or 8 (C/D row, A/B K+8)
    const int mrow   = mt * 16 + (lane & 15);   // A row for this lane
    const int col16  = lane & 15;

    // ---- Phase 2: GEMM1 [16x128]x[128x256] + GELU ----
    FragU a1[4];
    #pragma unroll
    for (int kt = 0; kt < 4; ++kt) {
        const unsigned short* p = &Xa[mrow * 128 + kt * 32 + rowoff];
        a1[kt].lo = *(const uint4*)p;           // K base..base+7
        a1[kt].hi = *(const uint4*)(p + 16);    // K base+16..base+23
    }
    #pragma unroll
    for (int ntile = 0; ntile < 16; ++ntile) {
        f32x8 c = {0.f, 0.f, 0.f, 0.f, 0.f, 0.f, 0.f, 0.f};
        #pragma unroll
        for (int kt = 0; kt < 4; ++kt) {
            const unsigned short* bp = w1t + ((long)(kt * 16 + ntile) * 32 + lane) * 16;
            FragU b;
            b.lo = *(const uint4*)bp;
            b.hi = *(const uint4*)(bp + 8);
            c = __builtin_amdgcn_wmma_f32_16x16x32_bf16(
                    false, as_bf(a1[kt]), false, as_bf(b), (short)0, c, false, false);
        }
        const int   col  = ntile * 16 + col16;
        const float bias = b1[col];
        #pragma unroll
        for (int r = 0; r < 8; ++r) {
            float gl = fast_gelu(c[r] + bias);
            Hb[(mt * 16 + rowoff + r) * 256 + col] = f2bf(gl);
        }
    }
    __syncthreads();

    // ---- Phase 3: GEMM2 [16x256]x[256x128] + bias + partial accum ----
    #pragma unroll
    for (int ntile = 0; ntile < 8; ++ntile) {
        f32x8 c = {0.f, 0.f, 0.f, 0.f, 0.f, 0.f, 0.f, 0.f};
        #pragma unroll
        for (int kt = 0; kt < 8; ++kt) {
            const unsigned short* p = &Hb[mrow * 256 + kt * 32 + rowoff];
            FragU a;
            a.lo = *(const uint4*)p;
            a.hi = *(const uint4*)(p + 16);
            const unsigned short* bp = w2t + ((long)(kt * 8 + ntile) * 32 + lane) * 16;
            FragU b;
            b.lo = *(const uint4*)bp;
            b.hi = *(const uint4*)(bp + 8);
            c = __builtin_amdgcn_wmma_f32_16x16x32_bf16(
                    false, as_bf(a), false, as_bf(b), (short)0, c, false, false);
        }
        const int   col   = ntile * 16 + col16;
        const float bias2 = b2[col];
        #pragma unroll
        for (int r = 0; r < 8; ++r) {
            const long row = tokBase + mt * 16 + rowoff + r;
            const long idx = row * DDIM + col;
            const float vv = c[r] + bias2;
            if (accumulate) outp[idx] += vv;
            else            outp[idx]  = vv;
        }
    }
}

// =====================================================================
// Host orchestration (all launches on `stream`, graph-capture safe)
// =====================================================================
extern "C" void kernel_launch(void* const* d_in, const int* in_sizes, int n_in,
                              void* d_out, int out_size, void* d_ws, size_t ws_size,
                              hipStream_t stream)
{
    const float* emb  = (const float*)d_in[0];
    const float* w    = (const float*)d_in[1];
    const float* ln_g = (const float*)d_in[2];
    const float* ln_b = (const float*)d_in[3];
    const float* W1   = (const float*)d_in[4];
    const float* b1   = (const float*)d_in[5];
    const float* W2   = (const float*)d_in[6];
    const float* b2   = (const float*)d_in[7];
    float* out = (float*)d_out;

    const long   ntok = (long)in_sizes[0] / DDIM;        // B*T = 65536
    const size_t SLOT = (size_t)ntok * DDIM * sizeof(float);

    // workspace: 4 committed-block slots + bf16 fragment-tiled weights
    char* ws = (char*)d_ws;
    float* blk[5];
    for (int i = 1; i <= 4; ++i) blk[i] = (float*)(ws + (size_t)(i - 1) * SLOT);
    unsigned short* w1t = (unsigned short*)(ws + 4 * SLOT);
    unsigned short* w2t = w1t + (size_t)LNUM * 128 * 256;

    // one-time weight repack (f32 -> bf16 fragment tiling)
    {
        const long tot = (long)LNUM * 128 * 256;
        const int  thr = 256;
        const long gb  = (tot + thr - 1) / thr;
        prep_kernel<<<(unsigned)gb, thr, 0, stream>>>(W1, w1t, 128, 256, 16, tot);
        prep_kernel<<<(unsigned)gb, thr, 0, stream>>>(W2, w2t, 256, 128, 8,  tot);
    }

    for (int l = 0; l < LNUM; ++l) {
        const int k = l / 4;
        const float* s[5] = { emb, emb, emb, emb, emb };
        int n = 1;
        for (int i = 1; i <= k; ++i) s[n++] = blk[i];
        if (l % 4 != 0)              s[n++] = blk[k + 1];   // in-flight partial

        if (l < LNUM - 1) {
            // fused mix+LN+MLP; h never leaves the WGP
            fused_layer_kernel<<<(unsigned)(ntok / TILE_M), 256, 0, stream>>>(
                s[0], s[1], s[2], s[3], s[4], n,
                w + (long)l * DDIM,
                ln_g + (long)l * DDIM, ln_b + (long)l * DDIM,
                w1t + (long)l * 128 * 256, b1 + (long)l * 256,
                w2t + (long)l * 256 * 128, b2 + (long)l * DDIM,
                blk[k + 1], (l % 4 != 0) ? 1 : 0);
        } else {
            // layer 15: output is h itself; its MLP is dead code
            const long nb = (ntok + 7) / 8;
            mix_kernel<<<(unsigned)nb, 256, 0, stream>>>(
                s[0], s[1], s[2], s[3], s[4], n, w + (long)l * DDIM, out, ntok);
        }
    }
}
